// MAE_54408645706405
// MI455X (gfx1250) — matrix operand order repaired and checked by reading, not compiled, not run on hardware
//
// MAE forward for MI455X (gfx1250). All large GEMMs via v_wmma_f32_16x16x32_bf16
// (4 WMMAs/wave/K-step, 64x128 block tile), fp32 accumulation, bf16 staging.
// A-tile staged with GLOBAL_LOAD_ASYNC_TO_LDS_B128 when the builtin exists.
#include <hip/hip_runtime.h>
#include <hip/hip_bf16.h>
#include <math.h>
#include <stdint.h>

typedef __attribute__((ext_vector_type(16))) __bf16 v16bf;
typedef __attribute__((ext_vector_type(8)))  __bf16 v8bf;
typedef __attribute__((ext_vector_type(8)))  float  v8f;
typedef __attribute__((ext_vector_type(4)))  int    v4i_t;

// ---------------- constants ----------------
namespace cfg {
constexpr int B = 64, NP = 196, NU = 49, NM = 147;
constexpr int E = 1024, EH = 16, EQKV = 3072, EMLP = 4096;
constexpr int D = 512,  DH = 8,  DQKV = 1536, DMLP = 2048;
constexpr int PD = 768;                 // 16*16*3 patch dim
constexpr int ME = B * NU;              // 3136 encoder rows
constexpr int MD = B * NP;              // 12544 decoder rows
constexpr int MM = B * NM;              // 9408 masked rows
}

// ---------------- helpers ----------------
__device__ inline __bf16 f2bf(float f) {
  union { float f; unsigned u; } x; x.f = f;
  unsigned r = x.u + 0x7FFFu + ((x.u >> 16) & 1u);
  union { unsigned short s; __bf16 b; } y; y.s = (unsigned short)(r >> 16);
  return y.b;
}
// tanh-GELU via sigmoid identity: 0.5x(1+tanh(u)) = x * sigmoid(2u)
__device__ inline float gelu_tanh(float x) {
  float t = 1.5957691216057308f * x + 0.0713548162726f * x * x * x;  // 2u
  return x / (1.0f + __expf(-t));
}

// ---------------- async global->LDS copy (gfx1250) ----------------
#ifndef USE_ASYNC_COPY
#define USE_ASYNC_COPY 1
#endif
#if defined(__has_builtin)
#if __has_builtin(__builtin_amdgcn_global_load_async_to_lds_b128)
#define HAVE_ASYNC_B128 1
#endif
#endif

#if USE_ASYNC_COPY && defined(HAVE_ASYNC_B128)
typedef __attribute__((address_space(1))) v4i_t g_v4i;   // global int4
typedef __attribute__((address_space(3))) v4i_t l_v4i;   // LDS int4
#endif

__device__ inline void copy16_g2l(const void* g, void* l) {
#if USE_ASYNC_COPY && defined(HAVE_ASYNC_B128)
  __builtin_amdgcn_global_load_async_to_lds_b128(
      (g_v4i*)(uintptr_t)g, (l_v4i*)(unsigned)(uintptr_t)l, 0, 0);
#else
  *(v8bf*)l = *(const v8bf*)g;
#endif
}
__device__ inline void copy_wait() {
#if USE_ASYNC_COPY && defined(HAVE_ASYNC_B128)
  asm volatile("s_wait_asynccnt 0x0" ::: "memory");
#endif
}

// ---------------- elementwise cast ----------------
__global__ __launch_bounds__(256) void k_cast(const float* __restrict__ s,
                                              __bf16* __restrict__ d, size_t n) {
  size_t i = (size_t)blockIdx.x * 256 + threadIdx.x;
  if (i < n) d[i] = f2bf(s[i]);
}

__global__ void k_zero1(float* p) { p[0] = 0.f; }

// ---------------- patchify: (B,224,224,3) -> bf16 (B*196, 768) ----------------
__global__ __launch_bounds__(256) void k_patchify(const float* __restrict__ img,
                                                  __bf16* __restrict__ xp) {
  size_t i = (size_t)blockIdx.x * 256 + threadIdx.x;
  size_t total = (size_t)cfg::B * cfg::NP * cfg::PD;
  if (i >= total) return;
  int e = (int)(i % cfg::PD); size_t r = i / cfg::PD;
  int p = (int)(r % cfg::NP); int b = (int)(r / cfg::NP);
  int py = p / 14, px = p % 14;
  int rr = e / 48, rem = e % 48, cc = rem / 3, ch = rem % 3;
  size_t src = (((size_t)b * 224 + (py * 16 + rr)) * 224 + (px * 16 + cc)) * 3 + ch;
  xp[i] = f2bf(img[src]);
}

// ---------------- per-image bitonic argsort of 196 keys ----------------
__global__ __launch_bounds__(256) void k_argsort(const float* __restrict__ rnd,
                                                 int* __restrict__ order) {
  int b = blockIdx.x, t = threadIdx.x;
  __shared__ float key[256]; __shared__ int idx[256];
  key[t] = (t < cfg::NP) ? rnd[b * cfg::NP + t] : 3.0e38f;
  idx[t] = t;
  __syncthreads();
  for (int k = 2; k <= 256; k <<= 1)
    for (int j = k >> 1; j > 0; j >>= 1) {
      int ixj = t ^ j;
      if (ixj > t) {
        bool up = ((t & k) == 0);
        float ka = key[t], kb = key[ixj];
        if (up ? (ka > kb) : (ka < kb)) {
          int ia = idx[t];
          key[t] = kb; key[ixj] = ka; idx[t] = idx[ixj]; idx[ixj] = ia;
        }
      }
      __syncthreads();
    }
  if (t < cfg::NP) order[b * cfg::NP + t] = idx[t];
}

// ---------------- WMMA GEMM: out = act(A@W + bias) + resid ----------------
// A: bf16 [M,K] row-major, W: bf16 [K,N] row-major. M%64==0, N%128==0, K%32==0.
// Block 256 threads = 8 waves; block tile 64(M)x128(N); each wave owns 32x32
// (2x2 WMMA accumulators), K-step 32 -> 4x v_wmma_f32_16x16x32_bf16 per step.
#define TM 64
#define TN 128
#define TK 32
__global__ __launch_bounds__(256)
void k_gemm(const __bf16* __restrict__ A, const __bf16* __restrict__ W,
            const float* __restrict__ bias, const float* __restrict__ resid,
            float* __restrict__ outf, __bf16* __restrict__ outb,
            int M, int N, int K, int act) {
  __shared__ alignas(16) __bf16 As[TM][TK];   // row-major tile of A (4 KB)
  __shared__ alignas(16) __bf16 Bs[TN][TK];   // transposed tile of W (8 KB)
  const int tid = threadIdx.x;
  const int m0 = blockIdx.y * TM, n0 = blockIdx.x * TN;
  const int w = tid >> 5, lane = tid & 31;
  const int wm = (w >> 2) * 32;       // 0,32
  const int wn = (w & 3) * 32;        // 0,32,64,96

  const int ar = tid >> 2, ac = (tid & 3) * 8;       // A loader: 1x16B per thread
  const int bk = tid >> 3, bn = (tid & 7) * 16;      // W loader: 2x16B per thread

  v8f c00 = {}, c01 = {}, c10 = {}, c11 = {};

  for (int k0 = 0; k0 < K; k0 += TK) {
    // A tile: async DMA into LDS (ASYNCcnt) when available
    copy16_g2l(A + (size_t)(m0 + ar) * K + k0 + ac, &As[ar][ac]);
    // W tile: load 16 contiguous-in-n bf16, store transposed (n-major)
    {
      v8bf v0 = *(const v8bf*)(W + (size_t)(k0 + bk) * N + n0 + bn);
      v8bf v1 = *(const v8bf*)(W + (size_t)(k0 + bk) * N + n0 + bn + 8);
#pragma unroll
      for (int i = 0; i < 8; ++i) { Bs[bn + i][bk] = v0[i]; Bs[bn + 8 + i][bk] = v1[i]; }
    }
    if (k0 + TK < K) {   // prefetch next K tile (global_prefetch_b8)
      __builtin_prefetch(A + (size_t)(m0 + ar) * K + k0 + TK + ac, 0, 1);
      __builtin_prefetch(W + (size_t)(k0 + TK + bk) * N + n0 + bn, 0, 1);
    }
    copy_wait();
    __syncthreads();

    // A fragments (16x32 bf16 each): lane l<16 row m K 0..7 & 16..23; l>=16 K 8..15 & 24..31
    const int abase = (lane >> 4) * 8;
    v16bf a0, a1;
    {
      const int am = wm + (lane & 15);
      v8bf lo = *(const v8bf*)&As[am][abase];
      v8bf hi = *(const v8bf*)&As[am][abase + 16];
      v8bf lo1 = *(const v8bf*)&As[am + 16][abase];
      v8bf hi1 = *(const v8bf*)&As[am + 16][abase + 16];
#pragma unroll
      for (int i = 0; i < 8; ++i) {
        a0[i] = lo[i];  a0[i + 8] = hi[i];
        a1[i] = lo1[i]; a1[i + 8] = hi1[i];
      }
    }
    // B fragments (32x16 each): lane l column n=l&15, l<16 K 0..7 & 16..23; l>=16 K 8..15 & 24..31
    const int bbase = (lane >> 4) * 8;
    v16bf b0, b1;
    {
      const int bnl = wn + (lane & 15);
      v8bf lo = *(const v8bf*)&Bs[bnl][bbase];
      v8bf hi = *(const v8bf*)&Bs[bnl][bbase + 16];
      v8bf lo1 = *(const v8bf*)&Bs[bnl + 16][bbase];
      v8bf hi1 = *(const v8bf*)&Bs[bnl + 16][bbase + 16];
#pragma unroll
      for (int i = 0; i < 8; ++i) {
        b0[i] = lo[i];  b0[i + 8] = hi[i];
        b1[i] = lo1[i]; b1[i + 8] = hi1[i];
      }
    }

    c00 = __builtin_amdgcn_wmma_f32_16x16x32_bf16(false, a0, false, b0, (short)0, c00, false, false);
    c01 = __builtin_amdgcn_wmma_f32_16x16x32_bf16(false, a0, false, b1, (short)0, c01, false, false);
    c10 = __builtin_amdgcn_wmma_f32_16x16x32_bf16(false, a1, false, b0, (short)0, c10, false, false);
    c11 = __builtin_amdgcn_wmma_f32_16x16x32_bf16(false, a1, false, b1, (short)0, c11, false, false);
    __syncthreads();
  }

  // epilogue. C layout: VGPR i, lane l -> C[(l>>4)*8 + i][l&15]
  const int nl = lane & 15;
  const int mb = (lane >> 4) * 8;
#pragma unroll
  for (int mi = 0; mi < 2; ++mi) {
#pragma unroll
    for (int nj = 0; nj < 2; ++nj) {
      const v8f& c = mi ? (nj ? c11 : c10) : (nj ? c01 : c00);
#pragma unroll
      for (int i = 0; i < 8; ++i) {
        int m = m0 + wm + mi * 16 + mb + i;
        int n = n0 + wn + nj * 16 + nl;
        float v = c[i];
        if (bias)  v += bias[n];
        if (act)   v = gelu_tanh(v);
        if (resid) v += resid[(size_t)m * N + n];
        size_t o = (size_t)m * N + n;
        if (outf) outf[o] = v;
        if (outb) outb[o] = f2bf(v);
      }
    }
  }
}

// ---------------- LayerNorm (fp32 in, bf16 out) ----------------
__global__ __launch_bounds__(256)
void k_layernorm(const float* __restrict__ x, const float* __restrict__ s,
                 const float* __restrict__ bb, __bf16* __restrict__ y, int Dd) {
  int row = blockIdx.x, t = threadIdx.x;
  const float* xr = x + (size_t)row * Dd;
  __shared__ float red[256];
  float acc = 0.f;
  for (int i = t; i < Dd; i += 256) acc += xr[i];
  red[t] = acc; __syncthreads();
  for (int st = 128; st > 0; st >>= 1) { if (t < st) red[t] += red[t + st]; __syncthreads(); }
  float mean = red[0] / Dd;
  __syncthreads();
  acc = 0.f;
  for (int i = t; i < Dd; i += 256) { float d = xr[i] - mean; acc += d * d; }
  red[t] = acc; __syncthreads();
  for (int st = 128; st > 0; st >>= 1) { if (t < st) red[t] += red[t + st]; __syncthreads(); }
  float rstd = rsqrtf(red[0] / Dd + 1e-5f);
  __syncthreads();
  for (int i = t; i < Dd; i += 256)
    y[(size_t)row * Dd + i] = f2bf((xr[i] - mean) * rstd * s[i] + bb[i]);
}

// ---------------- attention: qkv fp32 [B*N, 3*IE] -> o bf16 [B*N, IE] ----------------
// block = 128 threads = 4 waves; wave handles one query row; head dim fixed 64.
__global__ __launch_bounds__(128)
void k_attention(const float* __restrict__ qkv, __bf16* __restrict__ o,
                 int N, int H, int IE) {
  int bh = blockIdx.x, b = bh / H, h = bh % H;
  int w = threadIdx.x >> 5, lane = threadIdx.x & 31;
  int q = blockIdx.y * 4 + w;
  __shared__ float sc[4][224];
  bool active = (q < N);
  float inv = 0.f;
  if (active) {
    const float* qp = qkv + (size_t)(b * N + q) * (3 * IE) + h * 64;
    for (int j = lane; j < N; j += 32) {
      const float* kp = qkv + (size_t)(b * N + j) * (3 * IE) + IE + h * 64;
      float acc = 0.f;
      for (int d = 0; d < 64; ++d) acc += qp[d] * kp[d];
      sc[w][j] = acc * 0.125f;   // 1/sqrt(64)
    }
    float mx = -3.0e38f;
    for (int j = lane; j < N; j += 32) mx = fmaxf(mx, sc[w][j]);
    for (int d = 16; d > 0; d >>= 1) mx = fmaxf(mx, __shfl_xor(mx, d, 32));
    float sum = 0.f;
    for (int j = lane; j < N; j += 32) { float e = __expf(sc[w][j] - mx); sc[w][j] = e; sum += e; }
    for (int d = 16; d > 0; d >>= 1) sum += __shfl_xor(sum, d, 32);
    inv = 1.f / sum;
  }
  __syncthreads();           // make exp'd scores visible across lanes
  if (active) {
    float a0 = 0.f, a1 = 0.f;
    for (int j = 0; j < N; ++j) {
      float p = sc[w][j];
      const float* vp = qkv + (size_t)(b * N + j) * (3 * IE) + 2 * IE + h * 64;
      a0 += p * vp[lane]; a1 += p * vp[lane + 32];
    }
    __bf16* op = o + (size_t)(b * N + q) * IE + h * 64;
    op[lane] = f2bf(a0 * inv); op[lane + 32] = f2bf(a1 * inv);
  }
}

// ---------------- gathers / scatters ----------------
__global__ __launch_bounds__(256)
void k_gather_unmasked(const float* __restrict__ patches, const float* __restrict__ pos,
                       const int* __restrict__ order, float* __restrict__ xenc) {
  size_t i = (size_t)blockIdx.x * 256 + threadIdx.x;
  size_t total = (size_t)cfg::ME * cfg::E;
  if (i >= total) return;
  int e = (int)(i & 1023); size_t r = i >> 10;
  int j = (int)(r % cfg::NU); int b = (int)(r / cfg::NU);
  int u = order[b * cfg::NP + cfg::NM + j];
  xenc[i] = patches[((size_t)(b * cfg::NP + u) << 10) + e] + pos[u * cfg::E + e];
}

__global__ __launch_bounds__(256)
void k_scatter_dec(const float* __restrict__ dec_tok, const float* __restrict__ mask_tok,
                   const float* __restrict__ dec_pos, const int* __restrict__ order,
                   float* __restrict__ xdec) {
  size_t i = (size_t)blockIdx.x * 256 + threadIdx.x;
  size_t total = (size_t)cfg::B * cfg::NP * cfg::D;
  if (i >= total) return;
  int e = (int)(i & 511); size_t r = i >> 9;
  int j = (int)(r % cfg::NP); int b = (int)(r / cfg::NP);
  int pos = order[b * cfg::NP + j];
  float v = dec_pos[pos * cfg::D + e];
  if (j < cfg::NM) v += mask_tok[e];
  else v += dec_tok[((size_t)(b * cfg::NU + (j - cfg::NM)) << 9) + e];
  xdec[((size_t)(b * cfg::NP + pos) << 9) + e] = v;
}

__global__ __launch_bounds__(256)
void k_gather_masked(const __bf16* __restrict__ dec, const int* __restrict__ order,
                     __bf16* __restrict__ dm) {
  size_t i = (size_t)blockIdx.x * 256 + threadIdx.x;
  size_t total = (size_t)cfg::MM * cfg::D;
  if (i >= total) return;
  int e = (int)(i & 511); size_t r = i >> 9;
  int j = (int)(r % cfg::NM); int b = (int)(r / cfg::NM);
  int m = order[b * cfg::NP + j];
  dm[i] = dec[((size_t)(b * cfg::NP + m) << 9) + e];
}

// ---------------- masked MSE reduction ----------------
__global__ __launch_bounds__(256)
void k_mse(const float* __restrict__ pred, const float* __restrict__ patches,
           const int* __restrict__ order, float* __restrict__ out, float invn) {
  size_t i = (size_t)blockIdx.x * 256 + threadIdx.x;
  size_t total = (size_t)cfg::MM * cfg::E;
  float d2 = 0.f;
  if (i < total) {
    int e = (int)(i & 1023); size_t r = i >> 10;
    int j = (int)(r % cfg::NM); int b = (int)(r / cfg::NM);
    int m = order[b * cfg::NP + j];
    float t = patches[((size_t)(b * cfg::NP + m) << 10) + e];
    float df = pred[i] - t;
    d2 = df * df;
  }
  __shared__ float red[256];
  red[threadIdx.x] = d2; __syncthreads();
  for (int st = 128; st > 0; st >>= 1) {
    if (threadIdx.x < st) red[threadIdx.x] += red[threadIdx.x + st];
    __syncthreads();
  }
  if (threadIdx.x == 0) atomicAdd(out, red[0] * invn);
}

// ---------------- host orchestration ----------------
extern "C" void kernel_launch(void* const* d_in, const int* in_sizes, int n_in,
                              void* d_out, int out_size, void* d_ws, size_t ws_size,
                              hipStream_t stream) {
  using namespace cfg;
  const float* images   = (const float*)d_in[0];
  const float* rnd      = (const float*)d_in[1];
  const float* patch_W  = (const float*)d_in[2];
  const float* patch_b  = (const float*)d_in[3];
  const float* pos_emb  = (const float*)d_in[4];
  const float* enc_ln1s = (const float*)d_in[5];
  const float* enc_ln1b = (const float*)d_in[6];
  const float* enc_Wqkv = (const float*)d_in[7];
  const float* enc_Wo   = (const float*)d_in[8];
  const float* enc_ln2s = (const float*)d_in[9];
  const float* enc_ln2b = (const float*)d_in[10];
  const float* enc_W1   = (const float*)d_in[11];
  const float* enc_b1   = (const float*)d_in[12];
  const float* enc_W2   = (const float*)d_in[13];
  const float* enc_b2   = (const float*)d_in[14];
  const float* enc_lnfs = (const float*)d_in[15];
  const float* enc_lnfb = (const float*)d_in[16];
  const float* e2d_W    = (const float*)d_in[17];
  const float* e2d_b    = (const float*)d_in[18];
  const float* mask_tok = (const float*)d_in[19];
  const float* dec_pos  = (const float*)d_in[20];
  const float* dec_ln1s = (const float*)d_in[21];
  const float* dec_ln1b = (const float*)d_in[22];
  const float* dec_Wqkv = (const float*)d_in[23];
  const float* dec_Wo   = (const float*)d_in[24];
  const float* dec_ln2s = (const float*)d_in[25];
  const float* dec_ln2b = (const float*)d_in[26];
  const float* dec_W1   = (const float*)d_in[27];
  const float* dec_b1   = (const float*)d_in[28];
  const float* dec_W2   = (const float*)d_in[29];
  const float* dec_b2   = (const float*)d_in[30];
  const float* dec_lnfs = (const float*)d_in[31];
  const float* dec_lnfb = (const float*)d_in[32];
  const float* pix_W    = (const float*)d_in[33];
  const float* pix_b    = (const float*)d_in[34];
  (void)in_sizes; (void)n_in; (void)out_size; (void)ws_size;

  // ---- workspace bump allocator ----
  char* ws = (char*)d_ws; size_t off = 0;
  auto alloc = [&](size_t bytes) -> void* {
    off = (off + 255) & ~(size_t)255;
    void* p = ws + off; off += bytes; return p;
  };
  // bf16 weights
  __bf16* Wp_bf    = (__bf16*)alloc((size_t)PD * E * 2);
  __bf16* eWqkv_bf = (__bf16*)alloc((size_t)2 * E * EQKV * 2);
  __bf16* eWo_bf   = (__bf16*)alloc((size_t)2 * E * E * 2);
  __bf16* eW1_bf   = (__bf16*)alloc((size_t)2 * E * EMLP * 2);
  __bf16* eW2_bf   = (__bf16*)alloc((size_t)2 * EMLP * E * 2);
  __bf16* e2d_bf   = (__bf16*)alloc((size_t)E * D * 2);
  __bf16* dWqkv_bf = (__bf16*)alloc((size_t)D * DQKV * 2);
  __bf16* dWo_bf   = (__bf16*)alloc((size_t)D * D * 2);
  __bf16* dW1_bf   = (__bf16*)alloc((size_t)D * DMLP * 2);
  __bf16* dW2_bf   = (__bf16*)alloc((size_t)DMLP * D * 2);
  __bf16* pixW_bf  = (__bf16*)alloc((size_t)D * E * 2);
  // activations
  __bf16* Xp      = (__bf16*)alloc((size_t)MD * PD * 2);
  float*  patches = (float*) alloc((size_t)MD * E * 4);
  int*    order   = (int*)   alloc((size_t)B * NP * 4);
  float*  xenc    = (float*) alloc((size_t)ME * E * 4);
  float*  xdec    = (float*) alloc((size_t)MD * D * 4);
  size_t ln_elems  = ((size_t)ME * E > (size_t)MD * D) ? (size_t)ME * E : (size_t)MD * D;
  __bf16* ln_bf   = (__bf16*)alloc(ln_elems * 2);
  size_t qkv_elems = ((size_t)ME * EQKV > (size_t)MD * DQKV) ? (size_t)ME * EQKV : (size_t)MD * DQKV;
  float*  qkv     = (float*) alloc(qkv_elems * 4);
  size_t o_elems   = ((size_t)ME * E > (size_t)MD * D) ? (size_t)ME * E : (size_t)MD * D;
  __bf16* obf     = (__bf16*)alloc(o_elems * 2);
  size_t h_elems   = ((size_t)ME * EMLP > (size_t)MD * DMLP) ? (size_t)ME * EMLP : (size_t)MD * DMLP;
  __bf16* mlp_h   = (__bf16*)alloc(h_elems * 2);
  float*  dec_tok = (float*) alloc((size_t)ME * D * 4);
  __bf16* dm_bf   = (__bf16*)alloc((size_t)MM * D * 2);
  float*  pred    = (float*) alloc((size_t)MM * E * 4);

  auto cast = [&](const float* s, __bf16* d, size_t n) {
    k_cast<<<(unsigned)((n + 255) / 256), 256, 0, stream>>>(s, d, n);
  };
  auto gemm = [&](const __bf16* A, const __bf16* W, const float* bias, const float* resid,
                  float* outf, __bf16* outb, int M, int N, int K, int act) {
    dim3 g(N / TN, M / TM);
    k_gemm<<<g, 256, 0, stream>>>(A, W, bias, resid, outf, outb, M, N, K, act);
  };

  k_zero1<<<1, 1, 0, stream>>>((float*)d_out);

  // weights -> bf16
  cast(patch_W, Wp_bf,    (size_t)PD * E);
  cast(enc_Wqkv, eWqkv_bf, (size_t)2 * E * EQKV);
  cast(enc_Wo,  eWo_bf,   (size_t)2 * E * E);
  cast(enc_W1,  eW1_bf,   (size_t)2 * E * EMLP);
  cast(enc_W2,  eW2_bf,   (size_t)2 * EMLP * E);
  cast(e2d_W,   e2d_bf,   (size_t)E * D);
  cast(dec_Wqkv, dWqkv_bf, (size_t)D * DQKV);
  cast(dec_Wo,  dWo_bf,   (size_t)D * D);
  cast(dec_W1,  dW1_bf,   (size_t)D * DMLP);
  cast(dec_W2,  dW2_bf,   (size_t)DMLP * D);
  cast(pix_W,   pixW_bf,  (size_t)D * E);

  // patchify + patch embed
  {
    size_t n = (size_t)MD * PD;
    k_patchify<<<(unsigned)((n + 255) / 256), 256, 0, stream>>>(images, Xp);
  }
  gemm(Xp, Wp_bf, patch_b, nullptr, patches, nullptr, MD, E, PD, 0);

  // masking
  k_argsort<<<B, 256, 0, stream>>>(rnd, order);
  {
    size_t n = (size_t)ME * E;
    k_gather_unmasked<<<(unsigned)((n + 255) / 256), 256, 0, stream>>>(patches, pos_emb, order, xenc);
  }

  // ---- encoder (2 layers, dim 1024, 16 heads, seq 49) ----
  for (int l = 0; l < 2; ++l) {
    k_layernorm<<<ME, 256, 0, stream>>>(xenc, enc_ln1s + l * E, enc_ln1b + l * E, ln_bf, E);
    gemm(ln_bf, eWqkv_bf + (size_t)l * E * EQKV, nullptr, nullptr, qkv, nullptr, ME, EQKV, E, 0);
    k_attention<<<dim3(B * EH, (NU + 3) / 4), 128, 0, stream>>>(qkv, obf, NU, EH, E);
    gemm(obf, eWo_bf + (size_t)l * E * E, nullptr, xenc, xenc, nullptr, ME, E, E, 0);
    k_layernorm<<<ME, 256, 0, stream>>>(xenc, enc_ln2s + l * E, enc_ln2b + l * E, ln_bf, E);
    gemm(ln_bf, eW1_bf + (size_t)l * E * EMLP, enc_b1 + l * EMLP, nullptr, nullptr, mlp_h, ME, EMLP, E, 1);
    gemm(mlp_h, eW2_bf + (size_t)l * EMLP * E, enc_b2 + l * E, xenc, xenc, nullptr, ME, E, EMLP, 0);
  }
  k_layernorm<<<ME, 256, 0, stream>>>(xenc, enc_lnfs, enc_lnfb, ln_bf, E);

  // encoder -> decoder
  gemm(ln_bf, e2d_bf, e2d_b, nullptr, dec_tok, nullptr, ME, D, E, 0);
  {
    size_t n = (size_t)MD * D;
    k_scatter_dec<<<(unsigned)((n + 255) / 256), 256, 0, stream>>>(dec_tok, mask_tok, dec_pos, order, xdec);
  }

  // ---- decoder (1 layer, dim 512, 8 heads, seq 196) ----
  k_layernorm<<<MD, 256, 0, stream>>>(xdec, dec_ln1s, dec_ln1b, ln_bf, D);
  gemm(ln_bf, dWqkv_bf, nullptr, nullptr, qkv, nullptr, MD, DQKV, D, 0);
  k_attention<<<dim3(B * DH, (NP + 3) / 4), 128, 0, stream>>>(qkv, obf, NP, DH, D);
  gemm(obf, dWo_bf, nullptr, xdec, xdec, nullptr, MD, D, D, 0);
  k_layernorm<<<MD, 256, 0, stream>>>(xdec, dec_ln2s, dec_ln2b, ln_bf, D);
  gemm(ln_bf, dW1_bf, dec_b1, nullptr, nullptr, mlp_h, MD, DMLP, D, 1);
  gemm(mlp_h, dW2_bf, dec_b2, xdec, xdec, nullptr, MD, D, DMLP, 0);
  k_layernorm<<<MD, 256, 0, stream>>>(xdec, dec_lnfs, dec_lnfb, ln_bf, D);

  // pixel head on masked tokens + MSE
  {
    size_t n = (size_t)MM * D;
    k_gather_masked<<<(unsigned)((n + 255) / 256), 256, 0, stream>>>(ln_bf, order, dm_bf);
  }
  gemm(dm_bf, pixW_bf, pix_b, nullptr, pred, nullptr, MM, E, D, 0);
  {
    size_t n = (size_t)MM * E;
    float invn = 1.0f / (float)n;
    k_mse<<<(unsigned)((n + 255) / 256), 256, 0, stream>>>(pred, patches, order, (float*)d_out, invn);
  }
}
